// Seq2Seq_35364760716045
// MI455X (gfx1250) — matrix operand order repaired and checked by reading, not compiled, not run on hardware
//
#include <hip/hip_runtime.h>
#include <hip/hip_bf16.h>

// ---------------------------------------------------------------------------
// Seq2Seq LSTM (B=2048, SRC=512, TRG=128, H=50) for gfx1250 (MI455X).
// Persistent workgroup per 16-batch-row tile; 13 waves, one 16-col gate tile
// per wave, per-step GEMM via v_wmma_f32_16x16x32_f16 (K padded to 64).
// Recurrent weights live in VGPRs for the whole kernel. src/trg slabs are
// staged into LDS once via global_load_async_to_lds_b128 (ASYNCcnt), so the
// 639-step recurrence never touches global memory except output stores.
// Activations use gfx1250 v_tanh_f32 when available.
// ---------------------------------------------------------------------------

typedef __attribute__((ext_vector_type(16))) _Float16 v16h;
typedef __attribute__((ext_vector_type(8)))  _Float16 v8h;
typedef __attribute__((ext_vector_type(8)))  float    v8f;

union V16H { v16h v; _Float16 e[16]; v8h h2[2]; };

#define BT      16      // batch rows per workgroup
#define HH      50      // hidden size
#define G4      200     // 4*H gate columns
#define KPAD    64      // K padded for two 16x16x32 WMMA k-steps
#define NTILES  13      // ceil(200/16)
#define GSTR    256     // gates LDS row stride (floats)
#define SRC_LEN 512
#define TRG_LEN 128
#define NTHR    (NTILES * 32)   // 416 threads = 13 wave32

#if __has_builtin(__builtin_amdgcn_tanhf)
__device__ __forceinline__ float tanh_hw(float x) { return __builtin_amdgcn_tanhf(x); }
#elif __has_builtin(__builtin_amdgcn_tanh_f32)
__device__ __forceinline__ float tanh_hw(float x) { return __builtin_amdgcn_tanh_f32(x); }
#else
__device__ __forceinline__ float tanh_hw(float x) {
    float e = __expf(2.0f * x);
    return 1.0f - 2.0f / (e + 1.0f);
}
#endif
__device__ __forceinline__ float sigmoid_hw(float x) {
    return __builtin_fmaf(0.5f, tanh_hw(0.5f * x), 0.5f);
}

// CDNA5 async global->LDS copy (ASYNCcnt); lds_off = low 32 bits of flat addr.
__device__ __forceinline__ void async_b128(unsigned lds_off, const void* gaddr) {
    asm volatile("global_load_async_to_lds_b128 %0, %1, off"
                 :: "v"(lds_off), "v"(gaddr));
}
__device__ __forceinline__ void wait_async0() {
    asm volatile("s_wait_asynccnt 0" ::: "memory");
}
__device__ __forceinline__ unsigned lds_off_of(const void* p) {
    return (unsigned)(unsigned long long)p;   // LDS aperture: addr[31:0] == LDS offset
}

__global__ __launch_bounds__(NTHR) void
seq2seq_lstm_kernel(const float* __restrict__ src,
                    const float* __restrict__ trg,
                    const unsigned char* __restrict__ tf_mask,
                    const float* __restrict__ encWih,
                    const float* __restrict__ encWhh,
                    const float* __restrict__ encB,
                    const float* __restrict__ decWih,
                    const float* __restrict__ decWhh,
                    const float* __restrict__ decB,
                    const float* __restrict__ fcW,
                    const float* __restrict__ fcB,
                    float* __restrict__ out)
{
    __shared__ __align__(16) float    s_src[BT * SRC_LEN];   // 32 KB input slab
    __shared__ __align__(16) float    s_trg[BT * TRG_LEN];   // 8 KB target slab
    __shared__ __align__(16) _Float16 hbuf[BT * KPAD];       // h in WMMA-A layout
    __shared__ float  cbuf[BT * HH];                         // cell state
    __shared__ float  gates[BT * GSTR];                      // gate accumulators
    __shared__ __align__(16) float s_wbe[HH * 8];            // enc {wih x4, b x4} per j
    __shared__ __align__(16) float s_wbd[HH * 8];            // dec {wih x4, b x4} per j
    __shared__ float  s_fcw[HH + 2];
    __shared__ float  s_fcb;
    __shared__ float  xcur[BT];                              // decoder feedback input
    __shared__ unsigned char s_tf[TRG_LEN];

    const int tid  = threadIdx.x;
    const int lane = tid & 31;
    const int wid  = tid >> 5;            // wave id 0..12 == gate tile id
    const int nloc = lane & 15;
    const int half = lane >> 4;
    const int b0   = blockIdx.x * BT;

    // ---- async-stage contiguous src/trg slabs into LDS (ASYNCcnt path) ----
    {
        const float* gsrc = src + (size_t)b0 * SRC_LEN;      // contiguous 32 KB
        const float* gtrg = trg + (size_t)b0 * TRG_LEN;      // contiguous 8 KB
        for (int i4 = tid; i4 < (BT * SRC_LEN) / 4; i4 += NTHR)
            async_b128(lds_off_of(&s_src[i4 * 4]), gsrc + i4 * 4);
        for (int i4 = tid; i4 < (BT * TRG_LEN) / 4; i4 += NTHR)
            async_b128(lds_off_of(&s_trg[i4 * 4]), gtrg + i4 * 4);
    }

    // ---- stage small params, zero state ----
    for (int i = tid; i < BT * KPAD; i += NTHR) hbuf[i] = (_Float16)0.0f;
    for (int i = tid; i < BT * HH;   i += NTHR) cbuf[i] = 0.0f;
    for (int i = tid; i < HH * 8; i += NTHR) {
        int j = i >> 3, q = i & 7;
        s_wbe[i] = (q < 4) ? encWih[j + 50 * q] : encB[j + 50 * (q - 4)];
        s_wbd[i] = (q < 4) ? decWih[j + 50 * q] : decB[j + 50 * (q - 4)];
    }
    if (tid < HH) s_fcw[tid] = fcW[tid];
    if (tid == 0) s_fcb = fcB[0];
    if (tid < TRG_LEN) s_tf[tid] = tf_mask[tid];
    if (tid < BT) out[(size_t)(b0 + tid) * TRG_LEN + 0] = 0.0f;   // outputs[:,0] = 0

    // ---- gather recurrent weights into VGPR B-fragments (one time) ----
    // B[k][n] = W_hh[n][k]; 16-bit B layout: lane=(n, halfgroup), elem e -> K = 8*half + (e&7) + 16*(e>>3) (+32*frag)
    auto loadB = [&](const float* __restrict__ W, int frag) -> v16h {
        V16H r;
        int n = wid * 16 + nloc;
#pragma unroll
        for (int e = 0; e < 16; ++e) {
            int k = frag * 32 + half * 8 + (e & 7) + ((e >> 3) << 4);
            float v = (n < G4 && k < HH) ? W[n * HH + k] : 0.0f;
            r.e[e] = (_Float16)v;
        }
        return r.v;
    };
    const v16h Be0 = loadB(encWhh, 0), Be1 = loadB(encWhh, 1);
    const v16h Bd0 = loadB(decWhh, 0), Bd1 = loadB(decWhh, 1);

    // ---- loop-invariant elementwise item descriptors (no div in the scan) ----
    const int m0 = tid / HH,           j0 = tid - m0 * HH;            // item 0: always valid
    const int i1 = tid + NTHR;
    const int m1 = i1 / HH,            j1 = i1 - m1 * HH;             // item 1: tid < 384
    const bool has1 = (i1 < BT * HH);  // cut at exact wave boundary (384 = 12*32)

    wait_async0();
    __syncthreads();

    // A fragment: lane row = nloc, elems 0..7 at K = 32f+8h, elems 8..15 at K = 32f+16+8h
    auto loadA = [&](int frag) -> v16h {
        V16H a;
        const _Float16* base = &hbuf[nloc * KPAD + frag * 32 + half * 8];
        a.h2[0] = *(const v8h*)(base);
        a.h2[1] = *(const v8h*)(base + 16);
        return a.v;
    };

    auto item = [&](int m, int j, const float* __restrict__ wb,
                    const float* __restrict__ xbase, int xs) {
        float x = xbase[m * xs];
        const float4 wv = *(const float4*)(wb + j * 8);
        const float4 bv = *(const float4*)(wb + j * 8 + 4);
        const float* g = &gates[m * GSTR + j];
        float gi = g[0]      + x * wv.x + bv.x;
        float gf = g[HH]     + x * wv.y + bv.y;
        float gg = g[2 * HH] + x * wv.z + bv.z;
        float go = g[3 * HH] + x * wv.w + bv.w;
        int   ci = m * HH + j;
        float c  = sigmoid_hw(gf) * cbuf[ci] + sigmoid_hw(gi) * tanh_hw(gg);
        float h  = sigmoid_hw(go) * tanh_hw(c);
        cbuf[ci] = c;
        hbuf[m * KPAD + j] = (_Float16)h;
    };

    // ---- one LSTM step ----
    auto lstm_step = [&](v16h B0, v16h B1, const float* __restrict__ wb,
                         const float* __restrict__ xbase, int xs) {
        v16h a0 = loadA(0);
        v16h a1 = loadA(1);
        v8f  d  = {};
        d = __builtin_amdgcn_wmma_f32_16x16x32_f16(false, a0, false, B0,
                                                   (short)0, d, false, false);
        d = __builtin_amdgcn_wmma_f32_16x16x32_f16(false, a1, false, B1,
                                                   (short)0, d, false, false);
#pragma unroll
        for (int r = 0; r < 8; ++r)                       // D: elem r -> M=r+8*half, N=nloc
            gates[(r + 8 * half) * GSTR + wid * 16 + nloc] = d[r];

        __syncthreads();                                   // gates ready

        item(m0, j0, wb, xbase, xs);
        if (has1) item(m1, j1, wb, xbase, xs);             // uniform for wave 12

        __syncthreads();                                   // h/c updated
    };

    // ---- encoder scan: x comes straight from the LDS src slab ----
    for (int t = 0; t < SRC_LEN; ++t)
        lstm_step(Be0, Be1, s_wbe, &s_src[t], SRC_LEN);

    // decoder inp0 = src[:, -1]
    if (tid < BT) xcur[tid] = s_src[tid * SRC_LEN + (SRC_LEN - 1)];

    // ---- decoder scan with teacher forcing ----
    for (int s = 0; s < TRG_LEN - 1; ++s) {
        lstm_step(Bd0, Bd1, s_wbd, xcur, 1);
        if (tid < BT) {
            int m = tid;
            float p = s_fcb;
#pragma unroll
            for (int j = 0; j < HH; ++j)
                p += (float)hbuf[m * KPAD + j] * s_fcw[j];
            out[(size_t)(b0 + m) * TRG_LEN + (s + 1)] = p;
            xcur[m] = (s_tf[s + 1] != 0) ? s_trg[m * TRG_LEN + (s + 1)] : p;
        }
        // next step's mid-barrier orders xcur for the elementwise readers
    }
}

extern "C" void kernel_launch(void* const* d_in, const int* in_sizes, int n_in,
                              void* d_out, int out_size, void* d_ws, size_t ws_size,
                              hipStream_t stream) {
    const float*         src    = (const float*)d_in[0];
    const float*         trg    = (const float*)d_in[1];
    const unsigned char* tfm    = (const unsigned char*)d_in[2];
    const float*         encWih = (const float*)d_in[3];
    const float*         encWhh = (const float*)d_in[4];
    const float*         encB   = (const float*)d_in[5];
    const float*         decWih = (const float*)d_in[6];
    const float*         decWhh = (const float*)d_in[7];
    const float*         decB   = (const float*)d_in[8];
    const float*         fcW    = (const float*)d_in[9];
    const float*         fcB    = (const float*)d_in[10];
    float*               out    = (float*)d_out;

    dim3 grid(2048 / BT);   // 128 persistent workgroups
    dim3 block(NTHR);       // 13 wave32 (one gate tile per wave)
    seq2seq_lstm_kernel<<<grid, block, 0, stream>>>(
        src, trg, tfm, encWih, encWhh, encB, decWih, decWhh, decB, fcW, fcB, out);
}
